// LatentActionIDM_79748952752321
// MI455X (gfx1250) — matrix operand order repaired and checked by reading, not compiled, not run on hardware
//
#include <hip/hip_runtime.h>
#include <hip/hip_bf16.h>
#include <math.h>

typedef __bf16 bf16_t;
typedef __attribute__((ext_vector_type(16))) __bf16 v16bf;
typedef __attribute__((ext_vector_type(8)))  __bf16 v8bf;
typedef __attribute__((ext_vector_type(8)))  float  v8f;

union BFragU { v16bf v; v8bf h[2]; };

#define NTOK   98304          // B*T = 32768*3
#define NCODES 1024
#define CDIM   64
#define DECAY  0.99f
#define BETA   0.25f
#define SMEPS  1e-5f

__device__ __forceinline__ float gelu_tanh(float x) {
  float x3 = x * x * x;
  return 0.5f * x * (1.0f + tanhf(0.7978845608028654f * (x + 0.044715f * x3)));
}

// Async DMA of 16 bytes from global into LDS (per lane). ASYNCcnt-tracked.
// LDS address = low 32 bits of the generic pointer (LDS aperture: ADDR[31:0] is the offset).
__device__ __forceinline__ void async_b128(unsigned lds_off, const bf16_t* g) {
#if defined(__AMDGCN__)
  asm volatile("global_load_async_to_lds_b128 %0, %1, off"
               :: "v"(lds_off), "v"(g) : "memory");
#endif
}
__device__ __forceinline__ unsigned lds_off32(const void* p) {
  return (unsigned)(unsigned long long)p;
}

// A-fragment (16x32 bf16): lane<16 -> row=lane, K = [kb..kb+7],[kb+16..kb+23]
//                          lane>=16 -> row=lane-16, K = [kb+8..kb+15],[kb+24..kb+31]
__device__ __forceinline__ v16bf lds_frag_a(const bf16_t* row, int kb, int half) {
  BFragU f;
  f.h[0] = *(const v8bf*)(row + kb + half * 8);
  f.h[1] = *(const v8bf*)(row + kb + half * 8 + 16);
  return f.v;
}
// B-fragment (32x16 bf16), B staged transposed (N-major rows of K):
// lane<16 -> col=lane, K=[kb..kb+7],[kb+16..kb+23]; lane>=16 -> col=lane-16, K=[kb+8..kb+15],[kb+24..kb+31]
__device__ __forceinline__ v16bf lds_frag_b(const bf16_t* row, int kb, int half) {
  BFragU f;
  f.h[0] = *(const v8bf*)(row + kb + half * 8);
  f.h[1] = *(const v8bf*)(row + kb + half * 8 + 16);
  return f.v;
}

// ---------------- prep kernels ----------------
// 8 f32 -> 8 bf16 per thread (b128 in, b128 in, b128 out)
__global__ void cast_bf16_vec8(const float* __restrict__ in, bf16_t* __restrict__ out, int n8) {
  int i = blockIdx.x * 256 + threadIdx.x;
  if (i < n8) {
    const float4* p = (const float4*)in + (size_t)i * 2;
    float4 a = p[0], b = p[1];
    union { v8bf v; uint4 u; } r;
    r.v[0] = (bf16_t)a.x; r.v[1] = (bf16_t)a.y; r.v[2] = (bf16_t)a.z; r.v[3] = (bf16_t)a.w;
    r.v[4] = (bf16_t)b.x; r.v[5] = (bf16_t)b.y; r.v[6] = (bf16_t)b.z; r.v[7] = (bf16_t)b.w;
    ((uint4*)out)[i] = r.u;
  }
}

// W is (K x N) row-major f32; Wt is (N x K) row-major bf16
__global__ void transpose_cast_kernel(const float* __restrict__ W, bf16_t* __restrict__ Wt,
                                      int K, int N) {
  int i = blockIdx.x * 256 + threadIdx.x;
  if (i < N * K) {
    int n = i / K, k = i % K;
    Wt[i] = (bf16_t)W[(size_t)k * N + n];
  }
}

// embeddings (64 x 1024) -> embT bf16 (1024 x 64) + per-code squared norm
__global__ void emb_prep_kernel(const float* __restrict__ emb, bf16_t* __restrict__ Et,
                                float* __restrict__ sq) {
  int k = blockIdx.x * 256 + threadIdx.x;
  if (k < NCODES) {
    float s = 0.0f;
    #pragma unroll
    for (int c = 0; c < CDIM; ++c) {
      float v = emb[(size_t)c * NCODES + k];
      Et[(size_t)k * CDIM + c] = (bf16_t)v;
      s += v * v;
    }
    sq[k] = s;
  }
}

__global__ void zero_kernel(float* __restrict__ p, int n) {
  int i = blockIdx.x * 256 + threadIdx.x;
  if (i < n) p[i] = 0.0f;
}

// ---------------- fused GEMM + bias + gelu (bf16 WMMA, async double-buffered) ------------
// Tile: 128(M) x 64(N), BK=64; 8 waves; wave w owns rows [16w,16w+16), all 64 cols.
// Tiles DMA'd into LDS with global_load_async_to_lds_b128, double buffered:
// prefetch tile t+1 while computing tile t; s_wait_asynccnt 6 keeps only next tile in flight.
template<int K, int N, bool F32_OUT>
__global__ __launch_bounds__(256)
void gemm_bias_gelu(const bf16_t* __restrict__ X,   // M x K
                    const bf16_t* __restrict__ Wt,  // N x K (pre-transposed)
                    const float*  __restrict__ bias,
                    bf16_t* __restrict__ Ybf,       // M x N
                    float*  __restrict__ Yf) {      // M x N (optional)
  constexpr int P = 72;                             // bf16 pitch: 144B (16B aligned, bank-spread)
  __shared__ __align__(16) bf16_t sA[2][128 * P];   // 2 x 18KB
  __shared__ __align__(16) bf16_t sB[2][64 * P];    // 2 x 9KB

  const int tid  = threadIdx.x;
  const int lane = tid & 31, wv = tid >> 5;
  const int l15  = lane & 15, lh = lane >> 4;
  const int m0 = blockIdx.x * 128;
  const int n0 = blockIdx.y * 64;

  const int rA = tid >> 1, cA = (tid & 1) * 32;     // A: 128 rows x 64 cols, 64B/thread
  const int rB = tid >> 2, cB = (tid & 3) * 16;     // B: 64 rows x 64 cols, 32B/thread

  v8f acc[4];
  #pragma unroll
  for (int j = 0; j < 4; ++j)
    #pragma unroll
    for (int r = 0; r < 8; ++r) acc[j][r] = 0.0f;

  auto prefetch = [&](int k0, int buf) {
    const bf16_t* gA = X + (size_t)(m0 + rA) * K + k0 + cA;
    unsigned la = lds_off32(&sA[buf][rA * P + cA]);
    async_b128(la +  0, gA +  0);
    async_b128(la + 16, gA +  8);
    async_b128(la + 32, gA + 16);
    async_b128(la + 48, gA + 24);
    const bf16_t* gB = Wt + (size_t)(n0 + rB) * K + k0 + cB;
    unsigned lb = lds_off32(&sB[buf][rB * P + cB]);
    async_b128(lb +  0, gB + 0);
    async_b128(lb + 16, gB + 8);
  };

  prefetch(0, 0);
  constexpr int TILES = K / 64;
  #pragma unroll
  for (int t = 0; t < TILES; ++t) {
    const int cur = t & 1;
    if (t + 1 < TILES) {
      prefetch((t + 1) * 64, cur ^ 1);
#if defined(__AMDGCN__)
      asm volatile("s_wait_asynccnt 0x6" ::: "memory");  // only next tile outstanding
#endif
    } else {
#if defined(__AMDGCN__)
      asm volatile("s_wait_asynccnt 0x0" ::: "memory");
#endif
    }
    __syncthreads();

    const bf16_t* a_row = &sA[cur][(wv * 16 + l15) * P];
    #pragma unroll
    for (int ks = 0; ks < 2; ++ks) {
      // load all fragments first, then issue 4 back-to-back WMMAs
      v16bf a  = lds_frag_a(a_row, ks * 32, lh);
      v16bf b0 = lds_frag_b(&sB[cur][(0 * 16 + l15) * P], ks * 32, lh);
      v16bf b1 = lds_frag_b(&sB[cur][(1 * 16 + l15) * P], ks * 32, lh);
      v16bf b2 = lds_frag_b(&sB[cur][(2 * 16 + l15) * P], ks * 32, lh);
      v16bf b3 = lds_frag_b(&sB[cur][(3 * 16 + l15) * P], ks * 32, lh);
      acc[0] = __builtin_amdgcn_wmma_f32_16x16x32_bf16(false, a, false, b0, (short)0, acc[0], false, false);
      acc[1] = __builtin_amdgcn_wmma_f32_16x16x32_bf16(false, a, false, b1, (short)0, acc[1], false, false);
      acc[2] = __builtin_amdgcn_wmma_f32_16x16x32_bf16(false, a, false, b2, (short)0, acc[2], false, false);
      acc[3] = __builtin_amdgcn_wmma_f32_16x16x32_bf16(false, a, false, b3, (short)0, acc[3], false, false);
    }
    __syncthreads();
  }

  // epilogue: D layout -> VGPR r: lanes0-15 (M=r,N=lane), lanes16-31 (M=r+8,N=lane-16)
  #pragma unroll
  for (int j = 0; j < 4; ++j) {
    const int n = n0 + j * 16 + l15;
    const float bv = bias[n];
    #pragma unroll
    for (int r = 0; r < 8; ++r) {
      const int m = m0 + wv * 16 + lh * 8 + r;
      float y = gelu_tanh(acc[j][r] + bv);
      Ybf[(size_t)m * N + n] = (bf16_t)y;
      if constexpr (F32_OUT) Yf[(size_t)m * N + n] = y;
    }
  }
}

// ---------------- VQ nearest-code search via WMMA ----------------
// argmin ||x-e||^2 == argmax (x.e - 0.5*||e||^2). 128 tokens/block, 16 code-chunks of 64.
__global__ __launch_bounds__(256)
void vq_argmax_kernel(const bf16_t* __restrict__ Xbf,  // NTOK x 64
                      const bf16_t* __restrict__ Et,   // 1024 x 64
                      const float*  __restrict__ sq,   // 1024
                      int*   __restrict__ idx_out,
                      float* __restrict__ idxf_out) {
  constexpr int PA = 72, PB = 72, PS = 65;             // 144B pitches (16B aligned)
  __shared__ __align__(16) bf16_t sA[128 * PA];
  __shared__ __align__(16) bf16_t sB[64 * PB];
  __shared__ float sS[128 * PS];

  const int tid  = threadIdx.x;
  const int lane = tid & 31, wv = tid >> 5;
  const int l15  = lane & 15, lh = lane >> 4;
  const int m0 = blockIdx.x * 128;

  { // stage full A tile once: 128 x 64 bf16 (16KB)
    int r = tid >> 1, cs = (tid & 1) * 32;
    const uint4* g = (const uint4*)(Xbf + (size_t)(m0 + r) * CDIM + cs);
    uint4* s = (uint4*)(sA + r * PA + cs);
    s[0] = g[0]; s[1] = g[1]; s[2] = g[2]; s[3] = g[3];
  }

  float best = -3.4e38f;
  int bidx = 0;

  for (int c = 0; c < NCODES / 64; ++c) {
    const int n0 = c * 64;
    { // stage 64 codes x 64 dims bf16 (8KB)
      int r = tid >> 2, cs = (tid & 3) * 16;
      const uint4* g = (const uint4*)(Et + (size_t)(n0 + r) * CDIM + cs);
      uint4* s = (uint4*)(sB + r * PB + cs);
      s[0] = g[0]; s[1] = g[1];
    }
    __syncthreads();

    #pragma unroll
    for (int j = 0; j < 4; ++j) {
      v8f acc;
      #pragma unroll
      for (int r = 0; r < 8; ++r) acc[r] = 0.0f;
      #pragma unroll
      for (int ks = 0; ks < 2; ++ks) {   // K = 64 = 2 x 32
        v16bf a = lds_frag_a(sA + (wv * 16 + l15) * PA, ks * 32, lh);
        v16bf b = lds_frag_b(sB + (j * 16 + l15) * PB, ks * 32, lh);
        acc = __builtin_amdgcn_wmma_f32_16x16x32_bf16(
            false, a, false, b, (short)0, acc, false, false);
      }
      #pragma unroll
      for (int r = 0; r < 8; ++r)
        sS[(wv * 16 + lh * 8 + r) * PS + j * 16 + l15] = acc[r];
    }
    __syncthreads();

    if (tid < 128) {   // first-max scan matches jnp.argmax tie-breaking
      for (int n = 0; n < 64; ++n) {
        float v = sS[tid * PS + n] - 0.5f * sq[n0 + n];
        if (v > best) { best = v; bidx = n0 + n; }
      }
    }
    __syncthreads();
  }

  if (tid < 128) {
    idx_out[m0 + tid]  = bidx;
    idxf_out[m0 + tid] = (float)bidx;
  }
}

// ---------------- gather + statistics (one-hot matmuls as scatter-adds) ----------------
__global__ __launch_bounds__(256)
void vq_gather_stats(const float* __restrict__ latent,  // NTOK x 64
                     const float* __restrict__ emb,     // 64 x 1024 (f32 original)
                     const int*   __restrict__ idx,
                     float* __restrict__ quant_out,     // NTOK x 64
                     float* __restrict__ dw,            // 64 x 1024
                     float* __restrict__ counts,        // 1024
                     float* __restrict__ sse) {
  __shared__ float red[256];
  const int n = blockIdx.x * 256 + threadIdx.x;
  float local = 0.0f;
  if (n < NTOK) {
    const int id = idx[n];
    atomicAdd(&counts[id], 1.0f);
    #pragma unroll
    for (int cidx = 0; cidx < CDIM; ++cidx) {
      float l = latent[(size_t)n * CDIM + cidx];
      float q = emb[(size_t)cidx * NCODES + id];
      quant_out[(size_t)n * CDIM + cidx] = q;   // straight-through: latent + sg(q-latent) == q
      float d = q - l;
      local += d * d;
      atomicAdd(&dw[(size_t)cidx * NCODES + id], l);
    }
  }
  red[threadIdx.x] = local;
  __syncthreads();
  for (int s = 128; s > 0; s >>= 1) {
    if (threadIdx.x < s) red[threadIdx.x] += red[threadIdx.x + s];
    __syncthreads();
  }
  if (threadIdx.x == 0) atomicAdd(sse, red[0]);
}

// ---------------- EMA update / loss / perplexity ----------------
__global__ __launch_bounds__(1024)
void vq_finalize(const float* __restrict__ counts,
                 const float* __restrict__ dw,
                 const float* __restrict__ sse,
                 const float* __restrict__ ema_ch,   // 1024
                 const float* __restrict__ ema_dw,   // 64 x 1024
                 float* __restrict__ out_loss,
                 float* __restrict__ out_perp,
                 float* __restrict__ out_newemb) {   // 64 x 1024
  __shared__ float red[1024];
  const int k = threadIdx.x;
  const float debias = 1.0f - powf(DECAY, 1001.0f);  // EMA_COUNTER + 1
  const float cnt = counts[k];
  const float clh = ema_ch[k] * DECAY + cnt * (1.0f - DECAY);
  const float csz = clh / debias;

  red[k] = csz;
  __syncthreads();
  for (int s = 512; s > 0; s >>= 1) {
    if (k < s) red[k] += red[k + s];
    __syncthreads();
  }
  const float nsum = red[0];
  __syncthreads();

  const float stable = (csz + SMEPS) / (nsum + (float)NCODES * SMEPS) * nsum;
  #pragma unroll 8
  for (int c = 0; c < CDIM; ++c) {
    float dwh = ema_dw[(size_t)c * NCODES + k] * DECAY +
                dw[(size_t)c * NCODES + k] * (1.0f - DECAY);
    out_newemb[(size_t)c * NCODES + k] = (dwh / debias) / stable;
  }

  const float p = cnt / (float)NTOK;
  red[k] = -p * logf(p + 1e-10f);
  __syncthreads();
  for (int s = 512; s > 0; s >>= 1) {
    if (k < s) red[k] += red[k + s];
    __syncthreads();
  }
  if (k == 0) {
    *out_perp = expf(red[0]);
    *out_loss = BETA * sse[0] / ((float)NTOK * (float)CDIM);
  }
}

// ---------------- orchestration ----------------
extern "C" void kernel_launch(void* const* d_in, const int* in_sizes, int n_in,
                              void* d_out, int out_size, void* d_ws, size_t ws_size,
                              hipStream_t stream) {
  const float* states = (const float*)d_in[0];
  const float* w_se   = (const float*)d_in[1];
  const float* b_se   = (const float*)d_in[2];
  const float* w0     = (const float*)d_in[3];
  const float* b0     = (const float*)d_in[4];
  const float* w1     = (const float*)d_in[5];
  const float* b1     = (const float*)d_in[6];
  const float* w2     = (const float*)d_in[7];
  const float* b2     = (const float*)d_in[8];
  const float* emb    = (const float*)d_in[9];
  const float* ema_ch = (const float*)d_in[10];
  const float* ema_dw = (const float*)d_in[11];

  char* ws = (char*)d_ws;
  size_t o = 0;
  bf16_t* states_bf = (bf16_t*)(ws + o); o += (size_t)NTOK * 256 * 2;
  bf16_t* x1        = (bf16_t*)(ws + o); o += (size_t)NTOK * 512 * 2;
  bf16_t* x2        = (bf16_t*)(ws + o); o += (size_t)NTOK * 512 * 2;
  float*  latf      = (float*) (ws + o); o += (size_t)NTOK * 64 * 4;
  bf16_t* latbf     = (bf16_t*)(ws + o); o += (size_t)NTOK * 64 * 2;
  bf16_t* wse_t     = (bf16_t*)(ws + o); o += (size_t)512 * 256 * 2;
  bf16_t* w0_t      = (bf16_t*)(ws + o); o += (size_t)512 * 512 * 2;
  bf16_t* w1_t      = (bf16_t*)(ws + o); o += (size_t)512 * 512 * 2;
  bf16_t* w2_t      = (bf16_t*)(ws + o); o += (size_t)64 * 512 * 2;
  bf16_t* emb_t     = (bf16_t*)(ws + o); o += (size_t)NCODES * 64 * 2;
  float*  emb_sq    = (float*) (ws + o); o += (size_t)NCODES * 4;
  int*    idx       = (int*)   (ws + o); o += (size_t)NTOK * 4;
  float*  dw        = (float*) (ws + o); o += (size_t)CDIM * NCODES * 4;  // contiguous with
  float*  counts    = (float*) (ws + o); o += (size_t)NCODES * 4;         // counts and
  float*  sse       = (float*) (ws + o); o += 256;                        // sse for one zero pass

  float* out = (float*)d_out;
  const size_t Q_OFF = 0;                                 // quantized (NTOK*64)
  const size_t LOSS_OFF = (size_t)NTOK * CDIM;            // 6291456
  const size_t PERP_OFF = LOSS_OFF + 1;
  const size_t IDX_OFF  = PERP_OFF + 1;                   // enc_idx (NTOK, as float)
  const size_t EMB_OFF  = IDX_OFF + NTOK;                 // new_embeddings (64*1024)

  // 0) zero the scatter accumulators (dw | counts | sse are contiguous)
  {
    int nz = CDIM * NCODES + NCODES + 64;
    zero_kernel<<<(nz + 255) / 256, 256, 0, stream>>>(dw, nz);
  }
  // 1) casts / transposes
  cast_bf16_vec8<<<(NTOK * 256 / 8 + 255) / 256, 256, 0, stream>>>(states, states_bf, NTOK * 256 / 8);
  transpose_cast_kernel<<<(512 * 256 + 255) / 256, 256, 0, stream>>>(w_se, wse_t, 256, 512);
  transpose_cast_kernel<<<(512 * 512 + 255) / 256, 256, 0, stream>>>(w0, w0_t, 512, 512);
  transpose_cast_kernel<<<(512 * 512 + 255) / 256, 256, 0, stream>>>(w1, w1_t, 512, 512);
  transpose_cast_kernel<<<(64 * 512 + 255) / 256, 256, 0, stream>>>(w2, w2_t, 512, 64);
  emb_prep_kernel<<<(NCODES + 255) / 256, 256, 0, stream>>>(emb, emb_t, emb_sq);

  // 2) MLP chain via bf16 WMMA (async double-buffered LDS pipeline)
  dim3 blk(256);
  gemm_bias_gelu<256, 512, false><<<dim3(NTOK / 128, 512 / 64), blk, 0, stream>>>(
      states_bf, wse_t, b_se, x1, nullptr);
  gemm_bias_gelu<512, 512, false><<<dim3(NTOK / 128, 512 / 64), blk, 0, stream>>>(
      x1, w0_t, b0, x2, nullptr);
  gemm_bias_gelu<512, 512, false><<<dim3(NTOK / 128, 512 / 64), blk, 0, stream>>>(
      x2, w1_t, b1, x1, nullptr);
  gemm_bias_gelu<512, 64, true><<<dim3(NTOK / 128, 1), blk, 0, stream>>>(
      x1, w2_t, b2, latbf, latf);

  // 3) nearest-code search (WMMA) + index outputs
  vq_argmax_kernel<<<NTOK / 128, blk, 0, stream>>>(latbf, emb_t, emb_sq, idx, out + IDX_OFF);

  // 4) gather + loss/statistics scatter
  vq_gather_stats<<<NTOK / 256, blk, 0, stream>>>(latf, emb, idx, out + Q_OFF, dw, counts, sse);

  // 5) EMA update + scalars
  vq_finalize<<<1, 1024, 0, stream>>>(counts, dw, sse, ema_ch, ema_dw,
                                      out + LOSS_OFF, out + PERP_OFF, out + EMB_OFF);
}
